// AttentionMechanism_12721693130827
// MI455X (gfx1250) — matrix operand order repaired and checked by reading, not compiled, not run on hardware
//
#include <hip/hip_runtime.h>
#include <hip/hip_bf16.h>
#include <math.h>

typedef __attribute__((ext_vector_type(16))) __bf16 v16bf;
typedef __attribute__((ext_vector_type(8)))  __bf16 v8bf;
typedef __attribute__((ext_vector_type(8)))  float  v8f;

#define LL 2
#define BB 32
#define SS 2048
#define DD 1024
#define KT (DD / 32)   // 32 K-tiles
#define FT (DD / 16)   // 64 N-tiles

// split fp32 into bf16 hi + bf16 lo (x ~= hi + lo, residual ~2^-17 rel)
__device__ __forceinline__ void bf16_split(float x, __bf16& hi, __bf16& lo) {
    __bf16 h = (__bf16)x;
    hi = h;
    lo = (__bf16)(x - (float)h);
}

// ---------------------------------------------------------------------------
// Kernel 0: split W_enc into bf16 hi/lo planes, pre-swizzled in WMMA
// B-fragment order: index = ((kt*FT + ft)*32 + lane)*16 + e
// where for B (32x16, K x N): lane = (kl/16)*16 + n, e = kl%16,
// kl = k%32, n = f%16. One-time cost over 1M elements.
// ---------------------------------------------------------------------------
__global__ __launch_bounds__(256)
void wsplit_kernel(const float* __restrict__ Wenc,
                   __bf16* __restrict__ whi, __bf16* __restrict__ wlo) {
    const int k   = blockIdx.x;          // 0..1023
    const int tid = threadIdx.x;
    const int kt  = k >> 5;
    const int kl  = k & 31;
    const int lh  = (kl >> 4) << 4;      // lane-half base
    const int e   = kl & 15;
    const float* row = Wenc + (size_t)k * DD;
    for (int f = tid; f < DD; f += 256) {
        const int ft   = f >> 4;
        const int n    = f & 15;
        const int lane = lh + n;
        const size_t idx = (((size_t)kt * FT + ft) * 32 + lane) * 16 + e;
        bf16_split(row[f], whi[idx], wlo[idx]);
    }
}

// ---------------------------------------------------------------------------
// Kernel 1: dec_proj[b,f] = sum_e dec_hidden[L-1,b,e] * W_dec[e,f] + b_dec[f]
// ---------------------------------------------------------------------------
__global__ __launch_bounds__(256)
void decproj_kernel(const float* __restrict__ dec_hidden,
                    const float* __restrict__ Wdec,
                    const float* __restrict__ b_dec,
                    float* __restrict__ dproj) {
    const int b   = blockIdx.x;
    const int tid = threadIdx.x;
    const float* h = dec_hidden + ((size_t)(LL - 1) * BB + b) * DD;  // last layer
    float a0 = 0.f, a1 = 0.f, a2 = 0.f, a3 = 0.f;
    for (int e = 0; e < DD; ++e) {
        const float hv = h[e];                       // uniform -> scalar load
        const float* wr = Wdec + (size_t)e * DD;
        a0 += hv * wr[tid];
        a1 += hv * wr[tid + 256];
        a2 += hv * wr[tid + 512];
        a3 += hv * wr[tid + 768];
    }
    float* out = dproj + (size_t)b * DD;
    out[tid]       = a0 + b_dec[tid];
    out[tid + 256] = a1 + b_dec[tid + 256];
    out[tid + 512] = a2 + b_dec[tid + 512];
    out[tid + 768] = a3 + b_dec[tid + 768];
}

// ---------------------------------------------------------------------------
// Kernel 2 (fused, WMMA bf16x3):
//   enc_proj tile = enc_out[b, s0:s0+16, :] @ W_enc         (fp32 via bf16 split)
//   scores[b,s]   = sum_f tanh(enc_proj + dec_proj[b,f] + b_enc[f]) * w_att[f]
//                   + b_att, masked.
// Grid: (B, S/16). Block: 256 threads = 8 waves; wave w owns columns
// [w*128, w*128+128) as 8 WMMA 16x16 accumulators.
// ---------------------------------------------------------------------------
__global__ __launch_bounds__(256)
void attn_scores_wmma(const float*  __restrict__ enc,     // [B,S,D]
                      const __bf16* __restrict__ whi,     // pre-swizzled (ws)
                      const __bf16* __restrict__ wlo,     // pre-swizzled (ws)
                      const float*  __restrict__ b_enc,   // [D]
                      const float*  __restrict__ dproj,   // [B,D] (ws)
                      const float*  __restrict__ w_att,   // [D]
                      const float*  __restrict__ b_att,   // [1]
                      const int*    __restrict__ mask,    // [B,S]
                      float* __restrict__ scores)         // [B,S] (ws)
{
    const int b    = blockIdx.x;
    const int s0   = blockIdx.y * 16;
    const int tid  = threadIdx.x;
    const int wave = tid >> 5;
    const int lane = tid & 31;
    const int half = lane >> 4;
    const int ln   = lane & 15;
    const int fbase = wave * 128;

    __shared__ __attribute__((aligned(16))) __bf16 sAhi[16 * 32];
    __shared__ __attribute__((aligned(16))) __bf16 sAlo[16 * 32];
    __shared__ float sScore[16];

    if (tid < 16) sScore[tid] = 0.0f;

    v8f acc[8];
#pragma unroll
    for (int t = 0; t < 8; ++t) acc[t] = (v8f)(0.0f);

    const float* encRow = enc + ((size_t)b * SS + s0) * DD;

    // Per-wave base into the pre-swizzled B planes (tile stride = 512 bf16).
    // Fragment (kt, ft=wave*8+t): offset = ((kt*FT + ft)*32 + lane)*16.
    const size_t laneOff = (size_t)lane * 16;

    for (int k0 = 0; k0 < DD; k0 += 32) {
        const int kt = k0 >> 5;
        __syncthreads();  // WAR on sA*, and covers sScore init on first iter
        // Cooperative A tile: 16 rows x 32 K fp32 -> bf16 hi/lo in LDS
        for (int i = tid; i < 512; i += 256) {
            const int m  = i >> 5;
            const int kk = i & 31;
            bf16_split(encRow[(size_t)m * DD + k0 + kk], sAhi[i], sAlo[i]);
        }
        __syncthreads();

        // A fragments (contiguous k order, same as the B planes). Per-lane halves are
        // contiguous 16B runs in the staging buffer -> vector LDS loads.
        // elements 0..15: K = half*16 + 0..15
        union V16 { v16bf v; v8bf h[2]; };
        V16 ua, ul;
        const v8bf* pah = (const v8bf*)(sAhi + ln * 32 + half * 16);
        const v8bf* pal = (const v8bf*)(sAlo + ln * 32 + half * 16);
        ua.h[0] = pah[0];
        ua.h[1] = pah[1];   // +8 bf16
        ul.h[0] = pal[0];
        ul.h[1] = pal[1];

        const size_t ktBase = ((size_t)kt * FT + (size_t)wave * 8) * 32 * 16;
        // prefetch next K-step's B block for this wave (8 tiles * 512 bf16)
        if (kt + 1 < KT) {
            const size_t nxt = (((size_t)(kt + 1)) * FT + (size_t)wave * 8) * 32 * 16;
            __builtin_prefetch(whi + nxt + laneOff, 0, 1);
            __builtin_prefetch(wlo + nxt + laneOff, 0, 1);
        }

#pragma unroll
        for (int t = 0; t < 8; ++t) {
            const size_t off = ktBase + (size_t)t * 512 + laneOff;
            const v16bf bhi = *(const v16bf*)(whi + off);
            const v16bf blo = *(const v16bf*)(wlo + off);
            // bf16x3: hi*hi + hi*lo + lo*hi, fp32 accumulate
            acc[t] = __builtin_amdgcn_wmma_f32_16x16x32_bf16(
                false, ua.v, false, bhi, (short)0, acc[t], false, false);
            acc[t] = __builtin_amdgcn_wmma_f32_16x16x32_bf16(
                false, ua.v, false, blo, (short)0, acc[t], false, false);
            acc[t] = __builtin_amdgcn_wmma_f32_16x16x32_bf16(
                false, ul.v, false, bhi, (short)0, acc[t], false, false);
        }
    }

    // Epilogue: acc[t][r] = enc_proj at row m = r + 8*half (s = s0+m),
    // col f = fbase + t*16 + ln. Fuse bias + dec_proj + tanh + w_att dot.
    float p[8];
#pragma unroll
    for (int r = 0; r < 8; ++r) p[r] = 0.0f;

#pragma unroll
    for (int t = 0; t < 8; ++t) {
        const int f = fbase + t * 16 + ln;
        const float dv = dproj[(size_t)b * DD + f];
        const float bv = b_enc[f];
        const float wv = w_att[f];
#pragma unroll
        for (int r = 0; r < 8; ++r) {
            p[r] += tanhf(acc[t][r] + dv + bv) * wv;
        }
    }
    // reduce across the 16 lanes of each half (xor < 16 stays within half)
#pragma unroll
    for (int mskl = 1; mskl < 16; mskl <<= 1) {
#pragma unroll
        for (int r = 0; r < 8; ++r) p[r] += __shfl_xor(p[r], mskl, 32);
    }
    if (ln == 0) {
#pragma unroll
        for (int r = 0; r < 8; ++r) atomicAdd(&sScore[r + 8 * half], p[r]);
    }
    __syncthreads();

    if (tid < 16) {
        const int s = s0 + tid;
        float sc = sScore[tid] + b_att[0];
        if (mask[(size_t)b * SS + s] == 0) sc = -1e10f;
        scores[(size_t)b * SS + s] = sc;
    }
}

// ---------------------------------------------------------------------------
// Kernel 3: softmax over S per batch -> attn
// ---------------------------------------------------------------------------
__global__ __launch_bounds__(256)
void softmax_kernel(const float* __restrict__ scores, float* __restrict__ attn) {
    const int b   = blockIdx.x;
    const int tid = threadIdx.x;
    __shared__ float red[256];
    const float* sc = scores + (size_t)b * SS;

    float mx = -3.4e38f;
    for (int s = tid; s < SS; s += 256) mx = fmaxf(mx, sc[s]);
    red[tid] = mx;
    __syncthreads();
    for (int o = 128; o > 0; o >>= 1) {
        if (tid < o) red[tid] = fmaxf(red[tid], red[tid + o]);
        __syncthreads();
    }
    mx = red[0];
    __syncthreads();

    float sum = 0.0f;
    for (int s = tid; s < SS; s += 256) sum += expf(sc[s] - mx);
    red[tid] = sum;
    __syncthreads();
    for (int o = 128; o > 0; o >>= 1) {
        if (tid < o) red[tid] += red[tid + o];
        __syncthreads();
    }
    const float inv = 1.0f / red[0];

    float* at = attn + (size_t)b * SS;
    for (int s = tid; s < SS; s += 256) at[s] = expf(sc[s] - mx) * inv;
}

// ---------------------------------------------------------------------------
// Kernel 4: context[b,f] = sum_s attn[b,s] * enc[b,s,f]   (streaming, mem-bound)
// ---------------------------------------------------------------------------
__global__ __launch_bounds__(256)
void context_kernel(const float* __restrict__ attn,
                    const float* __restrict__ enc,
                    float* __restrict__ ctx) {
    const int b   = blockIdx.x;
    const int tid = threadIdx.x;
    const float* e  = enc  + (size_t)b * SS * DD;
    const float* at = attn + (size_t)b * SS;
    float a0 = 0.f, a1 = 0.f, a2 = 0.f, a3 = 0.f;
    for (int s = 0; s < SS; ++s) {
        const float a = at[s];  // uniform -> scalar load
        const float* er = e + (size_t)s * DD;
        a0 += a * er[tid];
        a1 += a * er[tid + 256];
        a2 += a * er[tid + 512];
        a3 += a * er[tid + 768];
    }
    float* o = ctx + (size_t)b * DD;
    o[tid]       = a0;
    o[tid + 256] = a1;
    o[tid + 512] = a2;
    o[tid + 768] = a3;
}

// ---------------------------------------------------------------------------
extern "C" void kernel_launch(void* const* d_in, const int* in_sizes, int n_in,
                              void* d_out, int out_size, void* d_ws, size_t ws_size,
                              hipStream_t stream) {
    const float* dec_hidden = (const float*)d_in[0];  // [L,B,D]
    const float* enc_out    = (const float*)d_in[1];  // [B,S,D]
    const int*   mask       = (const int*)  d_in[2];  // [B,S]
    const float* W_dec      = (const float*)d_in[3];  // [D,D]
    const float* b_dec      = (const float*)d_in[4];  // [D]
    const float* W_enc      = (const float*)d_in[5];  // [D,D]
    const float* b_enc      = (const float*)d_in[6];  // [D]
    const float* w_att      = (const float*)d_in[7];  // [D,1]
    const float* b_att      = (const float*)d_in[8];  // [1]

    float* ctx_out  = (float*)d_out;                   // [B,1,D] = 32768 floats
    float* attn_out = (float*)d_out + (size_t)BB * DD; // [B,S]  = 65536 floats

    // ws layout (floats): dproj [B*D], scores [B*S], then bf16 planes
    float*  dproj  = (float*)d_ws;                         // 32768 floats
    float*  scores = (float*)d_ws + (size_t)BB * DD;       // 65536 floats
    __bf16* whi    = (__bf16*)((float*)d_ws + (size_t)BB * DD + (size_t)BB * SS); // 1M bf16
    __bf16* wlo    = whi + (size_t)DD * DD;                                       // 1M bf16

    wsplit_kernel<<<dim3(DD), dim3(256), 0, stream>>>(W_enc, whi, wlo);

    decproj_kernel<<<dim3(BB), dim3(256), 0, stream>>>(dec_hidden, W_dec, b_dec, dproj);

    attn_scores_wmma<<<dim3(BB, SS / 16), dim3(256), 0, stream>>>(
        enc_out, whi, wlo, b_enc, dproj, w_att, b_att, mask, scores);

    softmax_kernel<<<dim3(BB), dim3(256), 0, stream>>>(scores, attn_out);

    context_kernel<<<dim3(BB), dim3(256), 0, stream>>>(attn_out, enc_out, ctx_out);
}
